// MultiheadAttention_62225486184705
// MI455X (gfx1250) — matrix operand order repaired and checked by reading, not compiled, not run on hardware
//
#include <hip/hip_runtime.h>
#include <stddef.h>

// ---------------------------------------------------------------------------
// MultiheadAttention forward for MI455X (gfx1250, wave32, WMMA bf16)
// B=4, S=2048, D=1024, H=16, DH=64
// ---------------------------------------------------------------------------

typedef __bf16 bf16;
typedef __attribute__((ext_vector_type(16))) __bf16 bf16x16;
typedef __attribute__((ext_vector_type(8)))  __bf16 bf16x8;
typedef __attribute__((ext_vector_type(4)))  __bf16 bf16x4;
typedef __attribute__((ext_vector_type(8)))  float  f32x8;
typedef __attribute__((ext_vector_type(4)))  int    i32x4;

constexpr int Bc  = 4;
constexpr int Sq  = 2048;
constexpr int Dm  = 1024;
constexpr int Hh  = 16;
constexpr int DHd = 64;

// ---- CDNA5 async global->LDS copy (ASYNCcnt path), guarded -----------------
#if __has_builtin(__builtin_amdgcn_global_load_async_to_lds_b128)
#define HAS_ASYNC_LDS 1
typedef __attribute__((address_space(1))) i32x4 gv4i;   // global int4
typedef __attribute__((address_space(3))) i32x4 lv4i;   // LDS int4
#else
#define HAS_ASYNC_LDS 0
#endif

__device__ __forceinline__ void cp_async_b128(const bf16* g, bf16* l) {
#if HAS_ASYNC_LDS
  __builtin_amdgcn_global_load_async_to_lds_b128((gv4i*)g, (lv4i*)l, 0, 0);
#else
  *(bf16x8*)l = *(const bf16x8*)g;
#endif
}

__device__ __forceinline__ void async_wait0() {
#if HAS_ASYNC_LDS
#if __has_builtin(__builtin_amdgcn_s_wait_asynccnt)
  __builtin_amdgcn_s_wait_asynccnt(0);
#else
  asm volatile("s_wait_asynccnt 0" ::: "memory");
#endif
#endif
}

// ---- fragment helpers (layouts per CDNA5 ISA 7.12.2, wave32) ---------------
__device__ __forceinline__ bf16x16 frag_zip(const bf16* p0, const bf16* p1) {
  union { bf16x16 v; bf16x8 h[2]; } u;
  u.h[0] = *(const bf16x8*)p0;
  u.h[1] = *(const bf16x8*)p1;
  return u.v;
}

// A-matrix 16x32 (MxK), row-major source with leading dim `ld`.
__device__ __forceinline__ bf16x16 load_a_frag(const bf16* tile, int ld, int lane) {
  int m  = lane & 15;
  int hi = lane >> 4;
  const bf16* r = tile + (size_t)m * ld;
  return frag_zip(r + hi * 8, r + 16 + hi * 8);
}

// B-matrix 32x16 (KxN); source rows are B columns, K contiguous per row.
__device__ __forceinline__ bf16x16 load_b_frag(const bf16* tile, int ld, int lane) {
  int n  = lane & 15;
  int hi = lane >> 4;
  const bf16* r = tile + (size_t)n * ld;
  return frag_zip(r + hi * 8, r + 16 + hi * 8);
}

// ---- f32 -> bf16 conversion ------------------------------------------------
__global__ __launch_bounds__(256) void cvt_kernel(const float* __restrict__ in,
                                                  bf16* __restrict__ out, int n) {
  int i = (blockIdx.x * 256 + threadIdx.x) * 4;
  if (i < n) {
    float4 f = *(const float4*)(in + i);
    bf16x4 o = {(bf16)f.x, (bf16)f.y, (bf16)f.z, (bf16)f.w};
    *(bf16x4*)(out + i) = o;
  }
}

// ---- GEMM with bias: C[M,N] = A[M,K] @ W[N,K]^T + bias ---------------------
// 128x128 block tile, 4 waves, 64x64 per wave (16 WMMA : 16 ds_load_b128),
// double-buffered LDS with async global->LDS DMA overlapped with WMMA.
// MODE 0: write f32 row-major [M,N]; MODE 1: write bf16 head-split [B,H,S,DH].
template <int MODE>
__global__ __launch_bounds__(128) void gemm_bias_kernel(
    const bf16* __restrict__ A, const bf16* __restrict__ W,
    const float* __restrict__ bias, float* __restrict__ outF,
    bf16* __restrict__ outH, int M, int N, int K) {
  __shared__ alignas(16) bf16 sA[2][128 * 32];
  __shared__ alignas(16) bf16 sW[2][128 * 32];

  const int tid  = threadIdx.x;
  const int lane = tid & 31;
  const int wave = tid >> 5;
  const int wy   = wave >> 1;   // 0..1 -> 64-row stripe
  const int wx   = wave & 1;    // 0..1 -> 64-col stripe
  const int m0   = blockIdx.y * 128;
  const int n0   = blockIdx.x * 128;

  auto stage = [&](int buf, int k0) {
#pragma unroll
    for (int i = 0; i < 4; i++) {
      int c   = tid + i * 128;       // 512 chunks of 8 halves per tile
      int row = c >> 2;
      int col = (c & 3) * 8;
      cp_async_b128(&A[(size_t)(m0 + row) * K + k0 + col],
                    &sA[buf][row * 32 + col]);
      cp_async_b128(&W[(size_t)(n0 + row) * K + k0 + col],
                    &sW[buf][row * 32 + col]);
    }
  };

  f32x8 acc[4][4] = {};

  stage(0, 0);
  async_wait0();
  __syncthreads();

  for (int k0 = 0; k0 < K; k0 += 32) {
    const int cur = (k0 >> 5) & 1;
    if (k0 + 32 < K) stage(cur ^ 1, k0 + 32);  // overlap DMA with WMMA

    bf16x16 af[4], wf[4];
#pragma unroll
    for (int i = 0; i < 4; i++)
      af[i] = load_a_frag(&sA[cur][(wy * 64 + i * 16) * 32], 32, lane);
#pragma unroll
    for (int u = 0; u < 4; u++)
      wf[u] = load_b_frag(&sW[cur][(wx * 64 + u * 16) * 32], 32, lane);

#pragma unroll
    for (int i = 0; i < 4; i++)
#pragma unroll
      for (int u = 0; u < 4; u++)
        acc[i][u] = __builtin_amdgcn_wmma_f32_16x16x32_bf16(
            false, af[i], false, wf[u], (short)0, acc[i][u], false, false);

    async_wait0();
    __syncthreads();
  }

  const int hi = lane >> 4;
  const int nn = lane & 15;
#pragma unroll
  for (int i = 0; i < 4; i++) {
#pragma unroll
    for (int u = 0; u < 4; u++) {
#pragma unroll
      for (int r = 0; r < 8; r++) {
        int row = m0 + wy * 64 + i * 16 + r + hi * 8;
        int col = n0 + wx * 64 + u * 16 + nn;
        float v = acc[i][u][r] + bias[col];
        if (MODE == 0) {
          outF[(size_t)row * N + col] = v;
        } else {
          int b = row / Sq, s = row % Sq;
          int h = col / DHd, dh = col % DHd;
          outH[(((size_t)b * Hh + h) * Sq + s) * DHd + dh] = (bf16)v;
        }
      }
    }
  }
}

// ---- softmax tile update (MASKED specialization keeps full tiles cheap) ----
template <bool MASKED>
__device__ __forceinline__ void tile_softmax(const f32x8* sc, int qrow0, int j0,
                                             int hi, int nn, float* mrow,
                                             float* alpha, float* p0, float* p1) {
#pragma unroll
  for (int r = 0; r < 8; r++) {
    float v0 = sc[0][r];
    float v1 = sc[1][r];
    if (MASKED) {
      int row = qrow0 + r + hi * 8;
      if (j0 + nn > row)      v0 = -10000.0f;
      if (j0 + 16 + nn > row) v1 = -10000.0f;
    }
    float mx = fmaxf(v0, v1);
#pragma unroll
    for (int d = 1; d < 16; d <<= 1) mx = fmaxf(mx, __shfl_xor(mx, d, 16));
    float newm = fmaxf(mrow[r], mx);
    alpha[r] = __expf(mrow[r] - newm);
    mrow[r]  = newm;
    p0[r] = __expf(v0 - newm);
    p1[r] = __expf(v1 - newm);
  }
}

// ---- causal flash attention (per (b,h), 128 q-rows per block) --------------
// Double-buffered K/V tiles; async K DMA and V global load for tile j+1 are
// issued before computing tile j, so both overlap the WMMAs.
// Qh/Kh/Vh: bf16 [B,H,S,DH]; Out: bf16 [B,S,D] (heads re-merged)
__global__ __launch_bounds__(256) void attn_kernel(
    const bf16* __restrict__ Qh, const bf16* __restrict__ Kh,
    const bf16* __restrict__ Vh, bf16* __restrict__ Out) {
  __shared__ alignas(16) bf16 sK[2][32 * 64];   // 32 keys x 64 dh, row-major
  __shared__ alignas(16) bf16 sVt[2][64 * 32];  // transposed: [dh][key]
  __shared__ alignas(16) bf16 sP[8][16 * 32];   // per-wave P relayout scratch

  const int tid   = threadIdx.x;
  const int lane  = tid & 31;
  const int wave  = tid >> 5;
  const int bh    = blockIdx.y;
  const int b     = bh / Hh;
  const int h     = bh % Hh;
  const int q0blk = blockIdx.x * 128;
  const int qrow0 = q0blk + wave * 16;          // this wave's 16 q rows
  const int hi    = lane >> 4;
  const int nn    = lane & 15;

  const bf16* Qb = Qh + ((size_t)bh * Sq + qrow0) * DHd;
  const bf16* Kb = Kh + (size_t)bh * Sq * DHd;
  const bf16* Vb = Vh + (size_t)bh * Sq * DHd;

  // staging coordinates (one b128 chunk per thread per tile)
  const int skey = tid >> 3;
  const int sc8  = (tid & 7) * 8;

  // Q fragments, with 1/sqrt(DH)=0.125 folded in (power of 2 -> exact in bf16)
  bf16x16 aQ[2];
  aQ[0] = load_a_frag(Qb + 0, DHd, lane);
  aQ[1] = load_a_frag(Qb + 32, DHd, lane);
#pragma unroll
  for (int i = 0; i < 16; i++) {
    aQ[0][i] = (bf16)((float)aQ[0][i] * 0.125f);
    aQ[1][i] = (bf16)((float)aQ[1][i] * 0.125f);
  }

  // all-ones B fragment: row-sums via WMMA (sum = P @ 1, replicated per col)
  bf16x16 ones;
#pragma unroll
  for (int i = 0; i < 16; i++) ones[i] = (bf16)1.0f;

  f32x8 o[4] = {};
  float mrow[8], lrow[8];
#pragma unroll
  for (int r = 0; r < 8; r++) { mrow[r] = -1e30f; lrow[r] = 0.0f; }

  const int ntiles = (q0blk + 128) / 32;  // causal: keys up to block's max row

  // prologue: stage tile 0 into buffer 0
  cp_async_b128(&Kb[(size_t)skey * DHd + sc8], &sK[0][skey * 64 + sc8]);
  {
    bf16x8 vv = *(const bf16x8*)&Vb[(size_t)skey * DHd + sc8];
#pragma unroll
    for (int i = 0; i < 8; i++) sVt[0][(sc8 + i) * 32 + skey] = vv[i];
  }
  async_wait0();
  __syncthreads();

  for (int jt = 0; jt < ntiles; jt++) {
    const int j0  = jt * 32;
    const int cur = jt & 1;
    const bool more = (jt + 1 < ntiles);

    bf16x8 vv;
    if (more) {  // issue next tile's K DMA + V load before computing this one
      cp_async_b128(&Kb[(size_t)(j0 + 32 + skey) * DHd + sc8],
                    &sK[cur ^ 1][skey * 64 + sc8]);
      vv = *(const bf16x8*)&Vb[(size_t)(j0 + 32 + skey) * DHd + sc8];
    }

    if (j0 <= qrow0 + 15) {  // wave-uniform: EXEC all-ones for WMMA
      // scores: S[16x32] = Qs[16x64] @ K_tile[32x64]^T (4 WMMAs)
      f32x8 sc[2] = {};
#pragma unroll
      for (int t = 0; t < 2; t++) {
        bf16x16 bK0 = load_b_frag(&sK[cur][(t * 16) * 64 + 0], 64, lane);
        bf16x16 bK1 = load_b_frag(&sK[cur][(t * 16) * 64 + 32], 64, lane);
        sc[t] = __builtin_amdgcn_wmma_f32_16x16x32_bf16(
            false, aQ[0], false, bK0, (short)0, sc[t], false, false);
        sc[t] = __builtin_amdgcn_wmma_f32_16x16x32_bf16(
            false, aQ[1], false, bK1, (short)0, sc[t], false, false);
      }

      float p0[8], p1[8], alpha[8];
      if (j0 + 31 <= qrow0) {  // uniform branch: tile fully unmasked
        tile_softmax<false>(sc, qrow0, j0, hi, nn, mrow, alpha, p0, p1);
      } else {
        tile_softmax<true>(sc, qrow0, j0, hi, nn, mrow, alpha, p0, p1);
      }

#pragma unroll
      for (int u = 0; u < 4; u++)
#pragma unroll
        for (int r = 0; r < 8; r++) o[u][r] *= alpha[r];

      // relayout P (C-frag) -> A-frag via per-wave LDS scratch
      bf16* myP = &sP[wave][0];
#pragma unroll
      for (int r = 0; r < 8; r++) {
        myP[(r + hi * 8) * 32 + nn]      = (bf16)p0[r];
        myP[(r + hi * 8) * 32 + 16 + nn] = (bf16)p1[r];
      }
      asm volatile("s_wait_dscnt 0" ::: "memory");
      bf16x16 aP = load_a_frag(myP, 32, lane);

      // row sums via ones-WMMA (replaces 32 ds_bpermute per tile)
      f32x8 srw = {};
      srw = __builtin_amdgcn_wmma_f32_16x16x32_bf16(
          false, aP, false, ones, (short)0, srw, false, false);
#pragma unroll
      for (int r = 0; r < 8; r++) lrow[r] = lrow[r] * alpha[r] + srw[r];

      // O += P[16x32] @ V_tile[32x64] (4 WMMAs, B from transposed V in LDS)
#pragma unroll
      for (int u = 0; u < 4; u++) {
        bf16x16 bV = load_b_frag(&sVt[cur][(u * 16) * 32], 32, lane);
        o[u] = __builtin_amdgcn_wmma_f32_16x16x32_bf16(
            false, aP, false, bV, (short)0, o[u], false, false);
      }
    }

    if (more) {  // transpose-store next V tile after compute (latency hidden)
#pragma unroll
      for (int i = 0; i < 8; i++) sVt[cur ^ 1][(sc8 + i) * 32 + skey] = vv[i];
    }
    async_wait0();
    __syncthreads();
  }

  // normalize and write merged-head bf16 output [B,S,D]
  float linv[8];
#pragma unroll
  for (int r = 0; r < 8; r++) linv[r] = 1.0f / lrow[r];
#pragma unroll
  for (int u = 0; u < 4; u++)
#pragma unroll
    for (int r = 0; r < 8; r++) {
      int srow = qrow0 + r + hi * 8;
      int dh   = u * 16 + nn;
      Out[((size_t)b * Sq + srow) * Dm + h * DHd + dh] =
          (bf16)(o[u][r] * linv[r]);
    }
}

// ---------------------------------------------------------------------------
extern "C" void kernel_launch(void* const* d_in, const int* in_sizes, int n_in,
                              void* d_out, int out_size, void* d_ws,
                              size_t ws_size, hipStream_t stream) {
  const float* q  = (const float*)d_in[0];
  const float* k  = (const float*)d_in[1];
  const float* v  = (const float*)d_in[2];
  // d_in[3] = causal mask (structure known, unused)
  const float* wq = (const float*)d_in[4];
  const float* bq = (const float*)d_in[5];
  const float* wk = (const float*)d_in[6];
  const float* bk = (const float*)d_in[7];
  const float* wv = (const float*)d_in[8];
  const float* bv = (const float*)d_in[9];
  const float* wo = (const float*)d_in[10];
  const float* bo = (const float*)d_in[11];
  float* out = (float*)d_out;

  const size_t MB = (size_t)Bc * Sq * Dm;  // 8,388,608
  const size_t WN = (size_t)Dm * Dm;       // 1,048,576

  bf16* p   = (bf16*)d_ws;
  bf16* qb  = p; p += MB;
  bf16* kb  = p; p += MB;
  bf16* vb  = p; p += MB;
  bf16* wqb = p; p += WN;
  bf16* wkb = p; p += WN;
  bf16* wvb = p; p += WN;
  bf16* wob = p; p += WN;
  bf16* Qh  = p; p += MB;
  bf16* Kh  = p; p += MB;
  bf16* Vh  = p; p += MB;
  bf16* ao  = p; p += MB;

  cvt_kernel<<<dim3((unsigned)(MB / 1024)), 256, 0, stream>>>(q, qb, (int)MB);
  cvt_kernel<<<dim3((unsigned)(MB / 1024)), 256, 0, stream>>>(k, kb, (int)MB);
  cvt_kernel<<<dim3((unsigned)(MB / 1024)), 256, 0, stream>>>(v, vb, (int)MB);
  cvt_kernel<<<dim3((unsigned)(WN / 1024)), 256, 0, stream>>>(wq, wqb, (int)WN);
  cvt_kernel<<<dim3((unsigned)(WN / 1024)), 256, 0, stream>>>(wk, wkb, (int)WN);
  cvt_kernel<<<dim3((unsigned)(WN / 1024)), 256, 0, stream>>>(wv, wvb, (int)WN);
  cvt_kernel<<<dim3((unsigned)(WN / 1024)), 256, 0, stream>>>(wo, wob, (int)WN);

  const int M = Bc * Sq;  // 8192
  dim3 ggrid(Dm / 128, M / 128);  // (8, 64)
  gemm_bias_kernel<1><<<ggrid, 128, 0, stream>>>(qb, wqb, bq, nullptr, Qh, M, Dm, Dm);
  gemm_bias_kernel<1><<<ggrid, 128, 0, stream>>>(kb, wkb, bk, nullptr, Kh, M, Dm, Dm);
  gemm_bias_kernel<1><<<ggrid, 128, 0, stream>>>(vb, wvb, bv, nullptr, Vh, M, Dm, Dm);

  attn_kernel<<<dim3(Sq / 128, Bc * Hh), 256, 0, stream>>>(Qh, Kh, Vh, ao);

  gemm_bias_kernel<0><<<ggrid, 128, 0, stream>>>(ao, wob, bo, out, nullptr, M, Dm, Dm);
}
